// MultiHeadAttention_764504179141
// MI455X (gfx1250) — compile-verified, matrix operand order checked by reading, not run on hardware
//
#include <hip/hip_runtime.h>
#include <hip/hip_bf16.h>

// ---------------------------------------------------------------------------
// Problem constants (from reference): N=8, S=2048, D=1024
//   alignment = (Q*K/sqrt(D)) @ Wa^T + ba        GEMM  M=16384 K=1024 N=1024
//   attn      = softmax over S (axis=1)
//   weighted  = attn * rowsum(V)
//   out       = weighted @ Wc^T + bc             GEMM  M=16384 K=1024 N=1024
// Strategy: bf16 WMMA (v_wmma_f32_16x16x32_bf16) for both GEMMs, fused
// online-softmax * vsum pass producing the second GEMM's bf16 A operand.
// ---------------------------------------------------------------------------

typedef __bf16 bf16_t;
typedef __attribute__((ext_vector_type(16))) __bf16 v16bf;
typedef __attribute__((ext_vector_type(8)))  __bf16 v8bf;
typedef __attribute__((ext_vector_type(4)))  __bf16 v4bf;
typedef __attribute__((ext_vector_type(8)))  float  v8f;

#define NB   8
#define SEQ  2048
#define DIM  1024
#define MROWS (NB * SEQ)          // 16384

// ---------------------------------------------------------------------------
// Prep: scaled = (q*k)*sc  -> bf16   (vectorized x4)
// ---------------------------------------------------------------------------
__global__ __launch_bounds__(256) void qk_scale_bf16(
    const float* __restrict__ q, const float* __restrict__ k,
    bf16_t* __restrict__ out, float sc, int n4)
{
    int i = blockIdx.x * 256 + threadIdx.x;
    if (i >= n4) return;
    float4 qv = ((const float4*)q)[i];
    float4 kv = ((const float4*)k)[i];
    v4bf r;
    r.x = (bf16_t)(qv.x * kv.x * sc);
    r.y = (bf16_t)(qv.y * kv.y * sc);
    r.z = (bf16_t)(qv.z * kv.z * sc);
    r.w = (bf16_t)(qv.w * kv.w * sc);
    ((v4bf*)out)[i] = r;
}

// ---------------------------------------------------------------------------
// Prep: f32 -> bf16 weight conversion (vectorized x4)
// ---------------------------------------------------------------------------
__global__ __launch_bounds__(256) void f32_to_bf16(
    const float* __restrict__ in, bf16_t* __restrict__ out, int n4)
{
    int i = blockIdx.x * 256 + threadIdx.x;
    if (i >= n4) return;
    float4 x = ((const float4*)in)[i];
    v4bf r;
    r.x = (bf16_t)x.x; r.y = (bf16_t)x.y; r.z = (bf16_t)x.z; r.w = (bf16_t)x.w;
    ((v4bf*)out)[i] = r;
}

// ---------------------------------------------------------------------------
// Prep: vsum[row] = sum_d value[row, d]   (one wave32 per row, 8 rows/block)
// ---------------------------------------------------------------------------
__global__ __launch_bounds__(256) void row_sum(
    const float* __restrict__ v, float* __restrict__ out)
{
    int row  = blockIdx.x * 8 + (threadIdx.x >> 5);
    int lane = threadIdx.x & 31;
    const float* p = v + (size_t)row * DIM;
    float s = 0.f;
    for (int i = lane; i < DIM; i += 32) s += p[i];
    #pragma unroll
    for (int off = 16; off > 0; off >>= 1) s += __shfl_down(s, off, 32);
    if (lane == 0) out[row] = s;
}

// ---------------------------------------------------------------------------
// WMMA GEMM:  out[M,N] = A[M,K](bf16) @ W[N,K]^T(bf16) + bias[N]   (f32 out)
// Block = 256 threads = 8 waves. Wave w computes rows [blk.x*128 + w*16, +16),
// cols [blk.y*64, +64) as four 16x16x32-bf16 accumulators. Per the CDNA5
// 16-bit operand layout: lane%16 = matrix row (A) / col-row of W (B);
// lane/16 picks the K-halves; the 8 K-pairs per VGPR-quad are contiguous in
// memory -> two b128 loads per operand per K-step.
// ---------------------------------------------------------------------------
__global__ __launch_bounds__(256) void gemm_bf16_wmma(
    const bf16_t* __restrict__ A,    // M x K row-major
    const bf16_t* __restrict__ W,    // N x K row-major
    const float*  __restrict__ bias, // N
    float* __restrict__ out,         // M x N
    int M, int K, int N)
{
    const int lane = threadIdx.x & 31;
    const int wave = threadIdx.x >> 5;
    const int m0   = blockIdx.x * 128 + wave * 16;
    const int e0   = blockIdx.y * 64;
    const int row  = lane & 15;   // M-row within A tile / N-col within B tile
    const int half = lane >> 4;   // selects K halves

    const bf16_t* Arow = A + (size_t)(m0 + row) * K;

    v8f acc[4];
    #pragma unroll
    for (int t = 0; t < 4; ++t)
        acc[t] = (v8f){0.f,0.f,0.f,0.f,0.f,0.f,0.f,0.f};

    for (int k0 = 0; k0 < K; k0 += 32) {
        union { v16bf v; v8bf h[2]; } ua;
        ua.h[0] = *(const v8bf*)(Arow + k0 + half * 8);
        ua.h[1] = *(const v8bf*)(Arow + k0 + 16 + half * 8);
        const v16bf a = ua.v;

        #pragma unroll
        for (int t = 0; t < 4; ++t) {
            const bf16_t* Wr = W + (size_t)(e0 + t * 16 + row) * K + k0 + half * 8;
            union { v16bf v; v8bf h[2]; } ub;
            ub.h[0] = *(const v8bf*)(Wr);
            ub.h[1] = *(const v8bf*)(Wr + 16);
            acc[t] = __builtin_amdgcn_wmma_f32_16x16x32_bf16(
                false, a, false, ub.v, (short)0, acc[t], false, false);
        }
    }

    // D layout: lane L holds column n = L%16; VGPR r holds M = (L/16)*8 + r
    #pragma unroll
    for (int t = 0; t < 4; ++t) {
        const int ecol = e0 + t * 16 + row;
        const float bv = bias[ecol];
        #pragma unroll
        for (int r = 0; r < 8; ++r) {
            const int m = m0 + half * 8 + r;
            out[(size_t)m * N + ecol] = acc[t][r] + bv;
        }
    }
}

// ---------------------------------------------------------------------------
// Fused: column softmax over S (stride-DIM columns) * vsum -> bf16.
// Block = 256 threads as 64 e-lanes (coalesced) x 4 s-strips; online max/sum,
// LDS combine of the 4 strip partials, second pass emits bf16 weighted A.
// grid = (NB, DIM/64)
// ---------------------------------------------------------------------------
__global__ __launch_bounds__(256) void softmax_weight_bf16(
    const float* __restrict__ align, const float* __restrict__ vsum,
    bf16_t* __restrict__ weighted)
{
    const int tx = threadIdx.x & 63;   // e offset (coalesced)
    const int ty = threadIdx.x >> 6;   // s strip 0..3
    const int n  = blockIdx.x;
    const int e  = blockIdx.y * 64 + tx;

    const float* col = align + (size_t)n * SEQ * DIM + e;

    float m = -__builtin_inff();
    float s = 0.f;
    for (int si = ty; si < SEQ; si += 4) {
        float x = col[(size_t)si * DIM];
        float mn = fmaxf(m, x);
        s = s * __expf(m - mn) + __expf(x - mn);
        m = mn;
    }

    __shared__ float smax[4][64];
    __shared__ float ssum[4][64];
    smax[ty][tx] = m; ssum[ty][tx] = s;
    __syncthreads();
    if (ty == 0) {
        float M = smax[0][tx], Ssum = ssum[0][tx];
        #pragma unroll
        for (int j = 1; j < 4; ++j) {
            float mj = smax[j][tx], sj = ssum[j][tx];
            float mn = fmaxf(M, mj);
            Ssum = Ssum * __expf(M - mn) + sj * __expf(mj - mn);
            M = mn;
        }
        smax[0][tx] = M; ssum[0][tx] = Ssum;
    }
    __syncthreads();
    const float M   = smax[0][tx];
    const float inv = 1.f / ssum[0][tx];

    bf16_t* wcol = weighted + (size_t)n * SEQ * DIM + e;
    const float* vs = vsum + n * SEQ;
    for (int si = ty; si < SEQ; si += 4) {
        float x = col[(size_t)si * DIM];
        wcol[(size_t)si * DIM] = (bf16_t)(__expf(x - M) * inv * vs[si]);
    }
}

// ---------------------------------------------------------------------------
// Launch
// ---------------------------------------------------------------------------
extern "C" void kernel_launch(void* const* d_in, const int* in_sizes, int n_in,
                              void* d_out, int out_size, void* d_ws, size_t ws_size,
                              hipStream_t stream) {
    const float* q  = (const float*)d_in[0];
    const float* k  = (const float*)d_in[1];
    const float* v  = (const float*)d_in[2];
    const float* Wa = (const float*)d_in[3];
    const float* ba = (const float*)d_in[4];
    const float* Wc = (const float*)d_in[5];
    const float* bc = (const float*)d_in[6];
    float* out = (float*)d_out;

    // Workspace layout (bytes):
    //   [0,   32MB)  sbuf   : bf16 staging (scaled, then reused for weighted)
    //   [32,  96MB)  albuf  : fp32 alignment
    //   [96,  98MB)  Wa16   : bf16 W_align
    //   [98, 100MB)  Wc16   : bf16 W_ctx
    //   [100MB, +64KB) vsum : fp32 row sums of V
    char* ws = (char*)d_ws;
    bf16_t* sbuf  = (bf16_t*)(ws);
    float*  albuf = (float*) (ws + (size_t)32  * 1024 * 1024);
    bf16_t* Wa16  = (bf16_t*)(ws + (size_t)96  * 1024 * 1024);
    bf16_t* Wc16  = (bf16_t*)(ws + (size_t)98  * 1024 * 1024);
    float*  vsum  = (float*) (ws + (size_t)100 * 1024 * 1024);

    const int total   = MROWS * DIM;        // 16,777,216
    const int total4  = total / 4;
    const int wtotal4 = (DIM * DIM) / 4;    // 262,144
    const float sc = 1.0f / sqrtf((float)DIM);

    // 1) scaled = Q*K/sqrt(D) -> bf16
    qk_scale_bf16<<<total4 / 256, 256, 0, stream>>>(q, k, sbuf, sc, total4);
    // 2) weights -> bf16
    f32_to_bf16<<<wtotal4 / 256, 256, 0, stream>>>(Wa, Wa16, wtotal4);
    f32_to_bf16<<<wtotal4 / 256, 256, 0, stream>>>(Wc, Wc16, wtotal4);
    // 3) vsum = rowsum(V)
    row_sum<<<MROWS / 8, 256, 0, stream>>>(v, vsum);
    // 4) alignment = scaled @ Wa^T + ba
    {
        dim3 grid(MROWS / 128, DIM / 64);
        gemm_bf16_wmma<<<grid, 256, 0, stream>>>(sbuf, Wa16, ba, albuf,
                                                 MROWS, DIM, DIM);
    }
    // 5) weighted = softmax_S(alignment) * vsum -> bf16 (reuse sbuf)
    {
        dim3 grid(NB, DIM / 64);
        softmax_weight_bf16<<<grid, 256, 0, stream>>>(albuf, vsum, sbuf);
    }
    // 6) out = weighted @ Wc^T + bc
    {
        dim3 grid(MROWS / 128, DIM / 64);
        gemm_bf16_wmma<<<grid, 256, 0, stream>>>(sbuf, Wc16, bc, out,
                                                 MROWS, DIM, DIM);
    }
}
